// Recurrence_31147102830987
// MI455X (gfx1250) — matrix operand order repaired and checked by reading, not compiled, not run on hardware
//
#include <hip/hip_runtime.h>
#include <hip/hip_bf16.h>

typedef __attribute__((ext_vector_type(16))) __bf16 v16bf;
typedef __attribute__((ext_vector_type(8)))  float  v8f;

#define T_STEPS 32
#define NBATCH  256
#define DD      8
#define SPATN   64
#define SSUB    8
#define HID     128
#define ACTN    16
#define PP      386
#define P3      1158
#define NPAD    1184   /* 3P padded to multiple of 32  */
#define KPAD    416    /* P  padded to multiple of 32  */
#define INW     521
#define OUTW    154
#define DHW     512
#define EPSF    1e-8f

// ---------------------------------------------------------------------------
// WMMA fragment loaders (wave32, bf16 16x16x32)
// A (16x32, MxK): lanes 0-15 row m hold K={0..7,16..23}; lanes 16-31 K={8..15,24..31}
// B (32x16, KxN): lane n (0-15) holds K={0..7,16..23}; lanes 16-31 K={8..15,24..31}
// Both are two contiguous 16B loads per lane from row-major (row, K) storage.
// ---------------------------------------------------------------------------
__device__ __forceinline__ v16bf load_fragA(const __bf16* rowp, int kk, int lane) {
    const __bf16* p = rowp + kk + ((lane >> 4) << 3);
    union { uint4 u[2]; v16bf v; } U;
    U.u[0] = *reinterpret_cast<const uint4*>(p);
    U.u[1] = *reinterpret_cast<const uint4*>(p + 16);
    return U.v;
}
__device__ __forceinline__ v16bf load_fragB(const __bf16* rowp, int kk, int lane) {
    const __bf16* p = rowp + kk + ((lane >> 4) << 3);
    union { uint4 u[2]; v16bf v; } U;
    U.u[0] = *reinterpret_cast<const uint4*>(p);
    U.u[1] = *reinterpret_cast<const uint4*>(p + 16);
    return U.v;
}

// ---------------------------------------------------------------------------
// Register-blocked single-wave WMMA GEMM: each wave computes a 32x32 C tile
// (2x2 register block -> 4 v_wmma per K-chunk, A/B frags reused twice each).
// C[M x Nc] = A[M x K] * Bt[Nc x K]^T ; grid = (M/32, Nc/32), block = 32.
// Branch-free epilogue via template flags: +bias, relu, f32 or bf16 store.
// ---------------------------------------------------------------------------
template <int K, bool HASBIAS, bool RELU, bool BF16OUT>
__global__ void wmma_gemm(const __bf16* __restrict__ A, const __bf16* __restrict__ Bt,
                          const float* __restrict__ bias, float* __restrict__ Cf,
                          __bf16* __restrict__ Cbf, int Nc) {
    const int lane = threadIdx.x;
    const int rsel = lane & 15;
    const __bf16* arow0 = A  + (size_t)(blockIdx.x * 32 + rsel) * K;
    const __bf16* arow1 = arow0 + (size_t)16 * K;
    const __bf16* brow0 = Bt + (size_t)(blockIdx.y * 32 + rsel) * K;
    const __bf16* brow1 = brow0 + (size_t)16 * K;
    v8f acc[2][2];
#pragma unroll
    for (int i = 0; i < 2; ++i)
#pragma unroll
        for (int j = 0; j < 2; ++j)
#pragma unroll
            for (int e = 0; e < 8; ++e) acc[i][j][e] = 0.0f;
#pragma unroll 4
    for (int kk = 0; kk < K; kk += 32) {
        v16bf a0 = load_fragA(arow0, kk, lane);
        v16bf a1 = load_fragA(arow1, kk, lane);
        v16bf b0 = load_fragB(brow0, kk, lane);
        v16bf b1 = load_fragB(brow1, kk, lane);
        acc[0][0] = __builtin_amdgcn_wmma_f32_16x16x32_bf16(false, a0, false, b0,
                                                            (short)0, acc[0][0], false, false);
        acc[0][1] = __builtin_amdgcn_wmma_f32_16x16x32_bf16(false, a0, false, b1,
                                                            (short)0, acc[0][1], false, false);
        acc[1][0] = __builtin_amdgcn_wmma_f32_16x16x32_bf16(false, a1, false, b0,
                                                            (short)0, acc[1][0], false, false);
        acc[1][1] = __builtin_amdgcn_wmma_f32_16x16x32_bf16(false, a1, false, b1,
                                                            (short)0, acc[1][1], false, false);
    }
    const int mhalf = (lane >> 4) << 3;
    float bv[2];
#pragma unroll
    for (int j = 0; j < 2; ++j)
        bv[j] = HASBIAS ? bias[blockIdx.y * 32 + j * 16 + rsel] : 0.0f;
#pragma unroll
    for (int i = 0; i < 2; ++i) {
#pragma unroll
        for (int j = 0; j < 2; ++j) {
            const int n = blockIdx.y * 32 + j * 16 + rsel;
            const int mb = blockIdx.x * 32 + i * 16 + mhalf;
#pragma unroll
            for (int e = 0; e < 8; ++e) {
                float v = acc[i][j][e] + bv[j];
                if (RELU) v = fmaxf(v, 0.0f);
                size_t offc = (size_t)(mb + e) * Nc + n;
                if (BF16OUT) Cbf[offc] = (__bf16)v;
                else         Cf[offc]  = v;
            }
        }
    }
}

// ------------------------- weight conversion kernels -----------------------
__global__ void k_w_pwih(const float* __restrict__ w, __bf16* __restrict__ o) {
    int i = blockIdx.x * blockDim.x + threadIdx.x;
    if (i >= NPAD * HID) return;
    int j = i / HID, k = i % HID;
    o[i] = (j < P3) ? (__bf16)w[j * HID + k] : (__bf16)0.0f;
}
__global__ void k_w_pwhh(const float* __restrict__ w, __bf16* __restrict__ o) {
    int i = blockIdx.x * blockDim.x + threadIdx.x;
    if (i >= NPAD * KPAD) return;
    int j = i / KPAD, k = i % KPAD;
    o[i] = (j < P3 && k < PP) ? (__bf16)w[j * PP + k] : (__bf16)0.0f;
}
__global__ void k_w_w2(const float* __restrict__ w, __bf16* __restrict__ o) {
    int i = blockIdx.x * blockDim.x + threadIdx.x;
    if (i >= 1024 * HID) return;
    int row = i / HID, h = i % HID;          // row = oc*8 + d
    int oc = row >> 3, d = row & 7;
    o[i] = (__bf16)w[oc * 1024 + h * 8 + d];
}
__global__ void k_w_convs(const float* __restrict__ w, __bf16* __restrict__ o) {
    int i = blockIdx.x * blockDim.x + threadIdx.x;
    if (i >= 2 * HID * HID) return;
    o[i] = (__bf16)w[i];                      // (l, out, in) already Bt layout
}
__global__ void k_w_linf(const float* __restrict__ w, __bf16* __restrict__ o) {
    int i = blockIdx.x * blockDim.x + threadIdx.x;
    if (i >= HID * 8192) return;
    int oc = i / 8192, k = i % 8192;          // k = sp*128 + c (activation order)
    int sp = k / HID, c = k % HID;
    o[i] = (__bf16)w[oc * 8192 + c * SPATN + sp];
}
__global__ void k_xbf(const float* __restrict__ inputs, const float* __restrict__ emb,
                      __bf16* __restrict__ o) {
    int i = blockIdx.x * blockDim.x + threadIdx.x;
    if (i >= SSUB * NBATCH * HID) return;
    int s = i / (NBATCH * HID), r = i % (NBATCH * HID);
    int n = r / HID, h = r % HID;
    int st = (int)inputs[(size_t)n * INW + DHW + s];   // subtasks0 (t = 0)
    o[i] = (__bf16)emb[st * HID + h];
}
__global__ void k_zero_hbf(__bf16* __restrict__ h) {
    int i = blockIdx.x * blockDim.x + threadIdx.x;
    if (i < NBATCH * KPAD) h[i] = (__bf16)0.0f;
}

// ------------------------------ GRU update ---------------------------------
__global__ void k_gru(const float* __restrict__ gi, const float* __restrict__ gh,
                      const float* __restrict__ pbih, const float* __restrict__ pbhh,
                      const float* __restrict__ Xprev, float* __restrict__ Xcur,
                      __bf16* __restrict__ hbf) {
    int i = blockIdx.x * blockDim.x + threadIdx.x;
    if (i >= NBATCH * PP) return;
    int n = i / PP, q = i % PP;
    const float* gin = gi + (size_t)n * NPAD;
    const float* ghn = gh + (size_t)n * NPAD;
    float ir = gin[q]          + pbih[q];
    float iz = gin[PP + q]     + pbih[PP + q];
    float ig = gin[2 * PP + q] + pbih[2 * PP + q];
    float hr = ghn[q]          + pbhh[q];
    float hz = ghn[PP + q]     + pbhh[PP + q];
    float hg = ghn[2 * PP + q] + pbhh[2 * PP + q];
    float r  = 1.0f / (1.0f + expf(-(ir + hr)));
    float z  = 1.0f / (1.0f + expf(-(iz + hz)));
    float g  = tanhf(ig + r * hg);
    float hp = Xprev ? Xprev[(size_t)n * PP + q] : 0.0f;
    float h2 = (1.0f - z) * g + z * hp;
    Xcur[(size_t)n * PP + q] = h2;
    hbf[(size_t)n * KPAD + q] = (__bf16)h2;
}

// -------------------------------- p init -----------------------------------
__global__ void k_pinit(const float* __restrict__ rnn_hxs, const float* __restrict__ X,
                        float* __restrict__ p) {
    int n = threadIdx.x;
    if (n >= NBATCH) return;
    const float* hx = rnn_hxs + (size_t)n * OUTW;
    bool z = true;
    for (int j = 0; j < OUTW; ++j) if (hx[j] != 0.0f) z = false;
    for (int s = 0; s < SSUB; ++s)
        p[n * SSUB + s] = z ? X[((size_t)s * NBATCH + n) * PP + 1] : hx[OUTW - SSUB + s];
}

// ---------------------- r = einsum('ns,nsh->nh', p, M) ---------------------
__global__ void k_r(const float* __restrict__ p, const float* __restrict__ X,
                    __bf16* __restrict__ rbf) {
    int i = blockIdx.x * blockDim.x + threadIdx.x;
    if (i >= NBATCH * HID) return;
    int n = i / HID, h = i % HID;
    float acc = 0.0f;
#pragma unroll
    for (int s = 0; s < SSUB; ++s)
        acc += p[n * SSUB + s] * X[((size_t)s * NBATCH + n) * PP + 2 + h];
    rbf[i] = (__bf16)acc;
}

// -------------- y0 = relu(sum_d G[n,c,d]*base[n,d,sp] + b0[c]) -------------
// stored as A1[(n*64+sp)][c] bf16 so the next GEMM is row-major contiguous.
__global__ void k_y0(int t, const float* __restrict__ inputs, const float* __restrict__ G,
                     const float* __restrict__ b0, __bf16* __restrict__ A1) {
    int i = blockIdx.x * blockDim.x + threadIdx.x;
    if (i >= NBATCH * HID * SPATN) return;
    int n = i / (HID * SPATN), r = i % (HID * SPATN);
    int c = r / SPATN, sp = r % SPATN;
    const float* Grow = G + (size_t)n * 1024 + c * 8;
    const float* brow = inputs + (size_t)t * NBATCH * INW + (size_t)n * INW;
    float acc = b0[c];
#pragma unroll
    for (int d = 0; d < DD; ++d) acc += Grow[d] * brow[d * SPATN + sp];
    acc = fmaxf(acc, 0.0f);
    A1[((size_t)n * SPATN + sp) * HID + c] = (__bf16)acc;
}

// --------------------------- per-step head kernel --------------------------
__global__ void k_head(int t, const float* __restrict__ sbuf, const float* __restrict__ X,
                       float* __restrict__ p, const float* __restrict__ inputs,
                       float* __restrict__ out,
                       const float* __restrict__ actor_w, const float* __restrict__ actor_b,
                       const float* __restrict__ critic_w, const float* __restrict__ critic_b,
                       const float* __restrict__ psi_w, const float* __restrict__ psi_b) {
    __shared__ float sh_s[HID], sh_e[HID], red[4][HID];
    __shared__ float sh_log[ACTN], sh_prob[ACTN], sh_p2[SSUB], sh_misc[2];
    __shared__ int sh_a;
    const int n = blockIdx.x, tid = threadIdx.x;
    sh_s[tid] = sbuf[(size_t)n * HID + tid];
    __syncthreads();
    if (tid < ACTN) {
        float acc = actor_b[tid];
        for (int h = 0; h < HID; ++h) acc += sh_s[h] * actor_w[tid * HID + h];
        sh_log[tid] = acc;
    }
    if (tid == ACTN) {
        float acc = critic_b[0];
        for (int h = 0; h < HID; ++h) acc += sh_s[h] * critic_w[h];
        sh_misc[0] = acc;
    }
    __syncthreads();
    if (tid == 0) {
        float mx = sh_log[0];
        for (int j = 1; j < ACTN; ++j) mx = fmaxf(mx, sh_log[j]);
        float sum = 0.0f;
        for (int j = 0; j < ACTN; ++j) { sh_prob[j] = expf(sh_log[j] - mx); sum += sh_prob[j]; }
        for (int j = 0; j < ACTN; ++j) sh_prob[j] /= sum;
        float araw = inputs[(size_t)t * NBATCH * INW + (size_t)n * INW + (INW - 1)];
        int a;
        if (araw < 0.0f) {              // deterministic fallback (never hit for these inputs)
            a = 0; float bv = sh_log[0];
            for (int j = 1; j < ACTN; ++j) if (sh_log[j] > bv) { bv = sh_log[j]; a = j; }
        } else a = (int)araw;
        sh_a = a;
    }
    __syncthreads();
    const int a = sh_a;
    {   // e = (s x onehot(a)) @ psi_w^T + psi_b  (per-n column-gathered matvec)
        float acc = psi_b[tid];
        const float* pw = psi_w + (size_t)tid * (HID * ACTN) + a;
        for (int h = 0; h < HID; ++h) acc += sh_s[h] * pw[h * ACTN];
        sh_e[tid] = acc;
        red[0][tid] = acc * acc;
    }
    __syncthreads();
    for (int st = 64; st > 0; st >>= 1) {
        if (tid < st) red[0][tid] += red[0][tid + st];
        __syncthreads();
    }
    if (tid == 0) sh_misc[1] = fmaxf(sqrtf(red[0][0]), EPSF);
    __syncthreads();
    const float ne = sh_misc[1];
    for (int s = 0; s < SSUB; ++s) {
        const float* xr = X + ((size_t)s * NBATCH + n) * PP;
        float e = sh_e[tid];
        float mm = xr[2 + HID + tid], mp = xr[2 + 2 * HID + tid];
        red[0][tid] = e * mp; red[1][tid] = e * mm;
        red[2][tid] = mp * mp; red[3][tid] = mm * mm;
        __syncthreads();
        for (int st = 64; st > 0; st >>= 1) {
            if (tid < st)
                for (int j = 0; j < 4; ++j) red[j][tid] += red[j][tid + st];
            __syncthreads();
        }
        if (tid == 0) {
            float np = fmaxf(sqrtf(red[2][0]), EPSF);
            float nm = fmaxf(sqrtf(red[3][0]), EPSF);
            float cosp = red[0][0] / (ne * np), cosm = red[1][0] / (ne * nm);
            float pv = p[n * SSUB + s] + xr[0] * cosp;
            pv -= pv * cosm;
            sh_p2[s] = pv;
        }
        __syncthreads();
    }
    float* orow = out + ((size_t)t * NBATCH + n) * OUTW;
    orow[18 + tid] = sh_s[tid];
    if (tid == 0) {
        float mx = sh_p2[0];
        for (int s = 1; s < SSUB; ++s) mx = fmaxf(mx, sh_p2[s]);
        float sum = 0.0f;
        for (int s = 0; s < SSUB; ++s) { sh_p2[s] = expf(sh_p2[s] - mx); sum += sh_p2[s]; }
        for (int s = 0; s < SSUB; ++s) {
            float pv = sh_p2[s] / sum;
            p[n * SSUB + s] = pv;
            orow[146 + s]   = pv;
        }
        orow[0]  = (float)a;
        orow[17] = sh_misc[0];
        for (int j = 0; j < ACTN; ++j) orow[1 + j] = sh_prob[j];
    }
}

// ---------------------------------------------------------------------------
extern "C" void kernel_launch(void* const* d_in, const int* in_sizes, int n_in,
                              void* d_out, int out_size, void* d_ws, size_t ws_size,
                              hipStream_t stream) {
    (void)in_sizes; (void)n_in; (void)out_size; (void)ws_size;
    const float* inputs   = (const float*)d_in[0];
    const float* rnn_hxs  = (const float*)d_in[1];
    const float* emb      = (const float*)d_in[2];
    const float* pwih     = (const float*)d_in[3];
    const float* pwhh     = (const float*)d_in[4];
    const float* pbih     = (const float*)d_in[5];
    const float* pbhh     = (const float*)d_in[6];
    const float* conv0_w  = (const float*)d_in[7];
    const float* conv0_b  = (const float*)d_in[8];
    const float* convs_w  = (const float*)d_in[9];
    const float* convs_b  = (const float*)d_in[10];
    const float* linf_w   = (const float*)d_in[11];
    const float* linf_b   = (const float*)d_in[12];
    const float* psi_w    = (const float*)d_in[13];
    const float* psi_b    = (const float*)d_in[14];
    const float* actor_w  = (const float*)d_in[15];
    const float* actor_b  = (const float*)d_in[16];
    const float* critic_w = (const float*)d_in[17];
    const float* critic_b = (const float*)d_in[18];
    float* out = (float*)d_out;

    char* ws = (char*)d_ws;
    size_t off = 0;
    auto take = [&](size_t bytes) -> char* {
        char* q = ws + off;
        off = (off + bytes + 255) & ~(size_t)255;
        return q;
    };
    __bf16* pwih_bf = (__bf16*)take((size_t)NPAD * HID * 2);
    __bf16* pwhh_bf = (__bf16*)take((size_t)NPAD * KPAD * 2);
    __bf16* w2bt    = (__bf16*)take((size_t)1024 * HID * 2);
    __bf16* convsbt = (__bf16*)take((size_t)2 * HID * HID * 2);
    __bf16* linfbt  = (__bf16*)take((size_t)HID * 8192 * 2);
    __bf16* xbf     = (__bf16*)take((size_t)SSUB * NBATCH * HID * 2);
    __bf16* hbf     = (__bf16*)take((size_t)NBATCH * KPAD * 2);
    float*  gi      = (float*)take((size_t)NBATCH * NPAD * 4);
    float*  gh      = (float*)take((size_t)NBATCH * NPAD * 4);
    float*  Xbuf    = (float*)take((size_t)SSUB * NBATCH * PP * 4);
    float*  pbuf    = (float*)take((size_t)NBATCH * SSUB * 4);
    __bf16* rbf     = (__bf16*)take((size_t)NBATCH * HID * 2);
    float*  G       = (float*)take((size_t)NBATCH * 1024 * 4);
    __bf16* A1      = (__bf16*)take((size_t)NBATCH * SPATN * HID * 2);
    __bf16* A2      = (__bf16*)take((size_t)NBATCH * SPATN * HID * 2);
    __bf16* A3      = (__bf16*)take((size_t)NBATCH * SPATN * HID * 2);
    float*  sbuf    = (float*)take((size_t)NBATCH * HID * 4);

    const int B = 256;
    // weight conversion / staging (deterministic, every call)
    k_w_pwih <<<(NPAD * HID + B - 1) / B, B, 0, stream>>>(pwih, pwih_bf);
    k_w_pwhh <<<(NPAD * KPAD + B - 1) / B, B, 0, stream>>>(pwhh, pwhh_bf);
    k_w_w2   <<<(1024 * HID + B - 1) / B, B, 0, stream>>>(conv0_w, w2bt);
    k_w_convs<<<(2 * HID * HID + B - 1) / B, B, 0, stream>>>(convs_w, convsbt);
    k_w_linf <<<(HID * 8192 + B - 1) / B, B, 0, stream>>>(linf_w, linfbt);
    k_xbf    <<<(SSUB * NBATCH * HID + B - 1) / B, B, 0, stream>>>(inputs, emb, xbf);
    k_zero_hbf<<<(NBATCH * KPAD + B - 1) / B, B, 0, stream>>>(hbf);

    // ------------------------------ GRU scan -------------------------------
    for (int s = 0; s < SSUB; ++s) {
        wmma_gemm<HID, false, false, false><<<dim3(NBATCH / 32, NPAD / 32), 32, 0, stream>>>(
            xbf + (size_t)s * NBATCH * HID, pwih_bf, nullptr, gi, nullptr, NPAD);
        wmma_gemm<KPAD, false, false, false><<<dim3(NBATCH / 32, NPAD / 32), 32, 0, stream>>>(
            hbf, pwhh_bf, nullptr, gh, nullptr, NPAD);
        const float* Xprev = (s == 0) ? nullptr : Xbuf + (size_t)(s - 1) * NBATCH * PP;
        k_gru<<<(NBATCH * PP + B - 1) / B, B, 0, stream>>>(
            gi, gh, pbih, pbhh, Xprev, Xbuf + (size_t)s * NBATCH * PP, hbf);
    }
    k_pinit<<<1, NBATCH, 0, stream>>>(rnn_hxs, Xbuf, pbuf);

    // ------------------------------ time scan ------------------------------
    for (int t = 0; t < T_STEPS; ++t) {
        k_r<<<(NBATCH * HID + B - 1) / B, B, 0, stream>>>(pbuf, Xbuf, rbf);
        wmma_gemm<HID, false, false, false><<<dim3(NBATCH / 32, 1024 / 32), 32, 0, stream>>>(
            rbf, w2bt, nullptr, G, nullptr, 1024);
        k_y0<<<(NBATCH * HID * SPATN + B - 1) / B, B, 0, stream>>>(
            t, inputs, G, conv0_b, A1);
        wmma_gemm<HID, true, true, true><<<dim3((NBATCH * SPATN) / 32, HID / 32), 32, 0, stream>>>(
            A1, convsbt, convs_b, nullptr, A2, HID);
        wmma_gemm<HID, true, true, true><<<dim3((NBATCH * SPATN) / 32, HID / 32), 32, 0, stream>>>(
            A2, convsbt + HID * HID, convs_b + HID, nullptr, A3, HID);
        wmma_gemm<8192, true, false, false><<<dim3(NBATCH / 32, HID / 32), 32, 0, stream>>>(
            A3, linfbt, linf_b, sbuf, nullptr, HID);
        k_head<<<NBATCH, HID, 0, stream>>>(t, sbuf, Xbuf, pbuf, inputs, out,
                                           actor_w, actor_b, critic_w, critic_b,
                                           psi_w, psi_b);
    }
}
